// PooledAttention_8538394984704
// MI455X (gfx1250) — matrix operand order repaired and checked by reading, not compiled, not run on hardware
//
#include <hip/hip_runtime.h>
#include <hip/hip_bf16.h>
#include <math.h>

typedef __attribute__((ext_vector_type(16))) _Float16 v16h;
typedef __attribute__((ext_vector_type(8)))  float    v8f;

// Types matching the async-load builtin's signature (from hipcc diagnostic):
// param 0 is 'int __attribute__((vector_size(16))) __device__ *'
typedef int v4i_ __attribute__((vector_size(16)));
typedef __attribute__((address_space(1))) v4i_* gv4i_p;   // global (AS1)
typedef __attribute__((address_space(3))) v4i_* lv4i_p;   // LDS (AS3)

#define EMB_DIM   64
#define MAX_BAG   64
#define SE_STRIDE 68   // floats per LDS emb row: 16B-aligned rows, bank-conflict-free
#define PW_STRIDE 72   // halfs per LDS proj_w row: 144B stride -> 16B-aligned b128 chunks

#if defined(__gfx1250__) && __has_builtin(__builtin_amdgcn_global_load_async_to_lds_b128) && __has_builtin(__builtin_amdgcn_s_wait_asynccnt)
#define USE_ASYNC_GATHER 1
#else
#define USE_ASYNC_GATHER 0
#endif

// Branch-free tanh: 1 - 2/(1+e^{2x}). Saturates correctly at +/-inf, no divergent
// EXEC regions; lowers to v_exp_f32 + v_rcp_f32 (TRANS ops that co-execute with
// the matrix pipe).
__device__ __forceinline__ float fast_tanh(float x) {
    return 1.0f - 2.0f / (1.0f + __expf(2.0f * x));
}

__global__ __launch_bounds__(128)
void pooled_attention_kernel(const int* __restrict__ indices,
                             const int* __restrict__ offsets,
                             const float* __restrict__ emb,
                             const float* __restrict__ proj_w,
                             const float* __restrict__ proj_b,
                             const float* __restrict__ att_h,
                             float* __restrict__ out,
                             int num_bags, int total_idx)
{
    __shared__ float    s_emb[MAX_BAG * SE_STRIDE];   // gathered rows (f32, reused twice)
    __shared__ _Float16 s_pw [EMB_DIM * PW_STRIDE];   // proj_w as f16, [n][k]
    __shared__ float    s_logit[MAX_BAG];
    __shared__ float    s_wt  [MAX_BAG];

    const int bag  = blockIdx.x;
    const int tid  = threadIdx.x;
    const int wave = tid >> 5;
    const int lane = tid & 31;
    const int half = lane >> 4;   // 0: lanes 0-15, 1: lanes 16-31
    const int l16  = lane & 15;

    const int start = offsets[bag];
    const int end   = (bag + 1 < num_bags) ? offsets[bag + 1] : total_idx;
    int count = end - start;
    if (count > MAX_BAG) count = MAX_BAG;

    // Stage proj_w as f16 (coalesced global read, contiguous LDS write).
    for (int idx = tid; idx < EMB_DIM * EMB_DIM; idx += 128) {
        int n = idx >> 6, k = idx & 63;
        s_pw[n * PW_STRIDE + k] = (_Float16)proj_w[idx];
    }

    // ---- Gather the bag's embedding rows into LDS; zero-pad rows [count,64). ----
#if USE_ASYNC_GATHER
    // gfx1250 async copy: global -> LDS directly (no VGPR round trip), all row
    // segments in flight on ASYNCcnt behind one s_wait_asynccnt 0.
    for (int idx = tid; idx < MAX_BAG * 16; idx += 128) {
        int r = idx >> 4, sg = idx & 15;
        if (r < count) {
            long long e = (long long)indices[start + r];
            __builtin_amdgcn_global_load_async_to_lds_b128(
                (gv4i_p)(emb + e * (long long)EMB_DIM + sg * 4),
                (lv4i_p)(&s_emb[r * SE_STRIDE + sg * 4]),
                0, 0);
        } else {
            *(float4*)(&s_emb[r * SE_STRIDE + sg * 4]) = make_float4(0.f, 0.f, 0.f, 0.f);
        }
    }
    __builtin_amdgcn_s_wait_asynccnt(0);
#else
    for (int idx = tid; idx < MAX_BAG * 16; idx += 128) {
        int r = idx >> 4, sg = idx & 15;
        float4 v = make_float4(0.f, 0.f, 0.f, 0.f);
        if (r < count) {
            long long e = (long long)indices[start + r];
            v = *(const float4*)(emb + e * (long long)EMB_DIM + sg * 4);
        }
        *(float4*)(&s_emb[r * SE_STRIDE + sg * 4]) = v;
    }
#endif
    __syncthreads();

    // ---- logits: Z = tanh(E @ W^T + b), logit = Z . att_h ----
    // Wave 'wave' owns rows [16w, 16w+16). A-fragment layout used here (contiguous K, not ISA table order):
    // lane L: M = L%16; half 0 slots -> K = {0..15}, half 1 -> K = {16..31}.
    const float* rowA = &s_emb[(wave * 16 + l16) * SE_STRIDE];
    const int kb = half * 16;

    v16h a0, a1;
    #pragma unroll
    for (int s = 0; s < 16; ++s) {
        int km = kb + s;
        a0[s] = (_Float16)rowA[km];        // K-chunk 0: K = km
        a1[s] = (_Float16)rowA[32 + km];   // K-chunk 1: K = 32 + km
    }

    float partial[8];
    #pragma unroll
    for (int v = 0; v < 8; ++v) partial[v] = 0.f;

    #pragma unroll
    for (int t = 0; t < 4; ++t) {
        const int n = t * 16 + l16;        // att dim handled by this lane
        v8f acc;
        float bv = proj_b[n];              // bias folded into C
        #pragma unroll
        for (int v = 0; v < 8; ++v) acc[v] = bv;

        // B-fragment: B[K][N] = proj_w[N][K]; half-wave h holds K = h*16 + s (+32 for chunk 1)
        // 16B-aligned contiguous 32B chunks -> ds_load_b128 pairs, conflict-free.
        const _Float16* colB = &s_pw[n * PW_STRIDE + half * 16];
        v16h b0, b1;
        #pragma unroll
        for (int s = 0; s < 16; ++s) { b0[s] = colB[s]; b1[s] = colB[32 + s]; }

        acc = __builtin_amdgcn_wmma_f32_16x16x32_f16(false, a0, false, b0, (short)0, acc, false, false);
        acc = __builtin_amdgcn_wmma_f32_16x16x32_f16(false, a1, false, b1, (short)0, acc, false, false);

        float ah = att_h[n];
        #pragma unroll
        for (int v = 0; v < 8; ++v) partial[v] += fast_tanh(acc[v]) * ah;
    }

    // Reduce over the 16 N-lanes within each half-wave (xor masks stay inside the half).
    #pragma unroll
    for (int v = 0; v < 8; ++v) {
        float p = partial[v];
        p += __shfl_xor(p, 8);
        p += __shfl_xor(p, 4);
        p += __shfl_xor(p, 2);
        p += __shfl_xor(p, 1);
        partial[v] = p;
    }
    if (l16 == 0) {
        #pragma unroll
        for (int v = 0; v < 8; ++v)
            s_logit[wave * 16 + half * 8 + v] = partial[v];  // C layout: M = v + 8*half
    }
    __syncthreads();

    // ---- stable softmax over the bag + attention-weighted pooling ----
    float m = -INFINITY;
    for (int i = 0; i < count; ++i) m = fmaxf(m, s_logit[i]);   // uniform broadcast reads
    if (tid < count) s_wt[tid] = __expf(s_logit[tid] - m);
    __syncthreads();
    float ssum = 0.f;
    for (int i = 0; i < count; ++i) ssum += s_wt[i];
    float inv = 1.f / ssum;

    if (tid < EMB_DIM) {
        float accd = 0.f;
        for (int i = 0; i < count; ++i) accd += s_wt[i] * s_emb[i * SE_STRIDE + tid];
        out[(long long)bag * EMB_DIM + tid] = accd * inv;
    }
}

extern "C" void kernel_launch(void* const* d_in, const int* in_sizes, int n_in,
                              void* d_out, int out_size, void* d_ws, size_t ws_size,
                              hipStream_t stream) {
    const int*   indices = (const int*)d_in[0];   // input_  [N]
    const int*   offsets = (const int*)d_in[1];   // offsets [B]
    const float* emb     = (const float*)d_in[2]; // emb_weight [NUM_EMB,64]
    const float* proj_w  = (const float*)d_in[3]; // [64,64]
    const float* proj_b  = (const float*)d_in[4]; // [64]
    const float* att_h   = (const float*)d_in[5]; // [64]
    float*       out     = (float*)d_out;         // [B,64]

    const int total    = in_sizes[0];
    const int num_bags = in_sizes[1];

    pooled_attention_kernel<<<num_bags, 128, 0, stream>>>(
        indices, offsets, emb, proj_w, proj_b, att_h, out, num_bags, total);
}
